// CascadeDynamicMixing_77051713290510
// MI455X (gfx1250) — matrix operand order repaired and checked by reading, not compiled, not run on hardware
//
#include <hip/hip_runtime.h>
#include <hip/hip_bf16.h>

typedef __attribute__((ext_vector_type(16))) __bf16 v16bf;
typedef __attribute__((ext_vector_type(8)))  __bf16 v8bf;
typedef __attribute__((ext_vector_type(8)))  float  v8f;
typedef __attribute__((ext_vector_type(4)))  float  v4f;

#define BQ     1200   // B*N
#define GG     4
#define CC     64
#define PP     288
#define OPTS   32
#define QD     256
#define ODIM   256
#define PWSZ   4096   // C*C
#define SPSZ   9216   // OPTS*P
#define GPARAM 13312  // PWSZ+SPSZ
#define NGEN   53248  // G*GPARAM
#define NBIAS  384    // G*(C+OPTS)
#define HFLAT  8192   // OPTS*G*C

#define MLD    80     // LDS stride for M^T (halves); 160B rows, 16B aligned
#define HLD    296    // LDS stride for h^T (halves); 592B rows, 16B aligned

#define KSPLIT 4      // K-split for the W_v GEMM

// ---------------- WMMA operand loaders (wave32, 16x16x32 bf16) ----------------
__device__ __forceinline__ v16bf load_a_bf(const __bf16* A, int lda) {
    int lane = threadIdx.x & 31;
    const __bf16* p = A + (lane & 15) * lda + ((lane & 16) ? 8 : 0);
    v8bf lo = *(const v8bf*)p;
    v8bf hi = *(const v8bf*)(p + 16);
    return __builtin_shufflevector(lo, hi, 0,1,2,3,4,5,6,7,8,9,10,11,12,13,14,15);
}
__device__ __forceinline__ v16bf load_a_f32(const float* A, int lda) {
    int lane = threadIdx.x & 31;
    const float* p = A + (lane & 15) * lda + ((lane & 16) ? 8 : 0);
    v4f f0 = *(const v4f*)p;
    v4f f1 = *(const v4f*)(p + 4);
    v4f f2 = *(const v4f*)(p + 16);
    v4f f3 = *(const v4f*)(p + 20);
    v16bf a;
#pragma unroll
    for (int i = 0; i < 4; ++i) {
        a[i]      = (__bf16)f0[i];
        a[4 + i]  = (__bf16)f1[i];
        a[8 + i]  = (__bf16)f2[i];
        a[12 + i] = (__bf16)f3[i];
    }
    return a;
}
// B tile from K-transposed storage Bt[col][k]; two 16B reads per lane (k = 8h.., 16+8h..).
__device__ __forceinline__ v16bf load_bt(const __bf16* Bt, int ldt) {
    int lane = threadIdx.x & 31;
    const __bf16* p = Bt + (lane & 15) * ldt + ((lane & 16) ? 8 : 0);
    v8bf lo = *(const v8bf*)p;
    v8bf hi = *(const v8bf*)(p + 16);
    return __builtin_shufflevector(lo, hi, 0,1,2,3,4,5,6,7,8,9,10,11,12,13,14,15);
}

#define WMMA_BF16(acc, a, b) \
    __builtin_amdgcn_wmma_f32_16x16x32_bf16(false, (a), false, (b), (short)0, (acc), false, false)

// ---------------- Stage 0: converts ----------------
__global__ void cvt_bf16_kernel(const float* __restrict__ in, __bf16* __restrict__ out, int n) {
    int i = blockIdx.x * blockDim.x + threadIdx.x;
    if (i < n) out[i] = (__bf16)in[i];
}
__global__ void cvt_bf16_T_kernel(const float* __restrict__ in, __bf16* __restrict__ out,
                                  int K, int N) {
    size_t idx = (size_t)blockIdx.x * blockDim.x + threadIdx.x;
    if (idx >= (size_t)K * N) return;
    int c = (int)(idx / K), k = (int)(idx % K);
    out[idx] = (__bf16)in[(size_t)k * N + c];
}

// ---------------- Stage 1: mix-bias GEMM (tiny, scalar) ----------------
__global__ void mixbias_kernel(const float* __restrict__ query, const float* __restrict__ Wb,
                               const float* __restrict__ bb, float* __restrict__ mb) {
    int idx = blockIdx.x * blockDim.x + threadIdx.x;
    if (idx >= BQ * NBIAS) return;
    int row = idx / NBIAS, col = idx % NBIAS;
    const float* q = query + row * QD;
    float acc = bb[col];
#pragma unroll 4
    for (int k = 0; k < QD; ++k) acc += q[k] * Wb[k * NBIAS + col];
    mb[idx] = acc;
}

// ---------------- Stage 2: params GEMM (1200 x 53248, K=256) ----------------
// 3x4 register-blocked: each wave -> 48x64 of output (12 WMMA tiles).
#define GMT 3
#define GNT 4
__global__ __launch_bounds__(256) void gen_gemm_kernel(const __bf16* __restrict__ qbf,
                                                       const __bf16* __restrict__ wgenT,
                                                       const float* __restrict__ bgen,
                                                       __bf16* __restrict__ params) {
    const int nblk = (NGEN / 16) / GNT;                // 832
    int id = blockIdx.x * 8 + (threadIdx.x >> 5);      // wave-job id
    int bm = id / nblk, bn = id % nblk;
    const __bf16* A  = qbf + (size_t)bm * GMT * 16 * QD;
    const __bf16* Bt = wgenT + (size_t)bn * GNT * 16 * QD;
    v8f acc[GMT][GNT] = {};
#pragma unroll
    for (int kk = 0; kk < QD / 32; ++kk) {
        v16bf a[GMT];
#pragma unroll
        for (int i = 0; i < GMT; ++i) a[i] = load_a_bf(A + i * 16 * QD + kk * 32, QD);
#pragma unroll
        for (int j = 0; j < GNT; ++j) {
            v16bf b = load_bt(Bt + j * 16 * QD + kk * 32, QD);
#pragma unroll
            for (int i = 0; i < GMT; ++i) acc[i][j] = WMMA_BF16(acc[i][j], a[i], b);
        }
    }
    int lane = threadIdx.x & 31;
    int cl = lane & 15, hi = lane >> 4;
#pragma unroll
    for (int j = 0; j < GNT; ++j) {
        int col = bn * GNT * 16 + j * 16 + cl;
        float bg = bgen[col];
        float scale = ((col % GPARAM) < PWSZ) ? 0.125f : 1.0f;  // 1/sqrt(C), C=64
#pragma unroll
        for (int i = 0; i < GMT; ++i) {
            int rbase = bm * GMT * 16 + i * 16 + 8 * hi;
#pragma unroll
            for (int v = 0; v < 8; ++v)
                params[(size_t)(rbase + v) * NGEN + col] = (__bf16)((acc[i][j][v] + bg) * scale);
        }
    }
}

// ---------------- Stage 3: dynamic mixing, one block per (b,n,g) ----------------
__global__ __launch_bounds__(256) void mix_kernel(const float* __restrict__ x,
                                                  const __bf16* __restrict__ params,
                                                  const float* __restrict__ mb,
                                                  __bf16* __restrict__ h2) {
    __shared__ __attribute__((aligned(16))) __bf16 Mt[CC * MLD];  // 10.0 KB  M^T[d][c]
    __shared__ __attribute__((aligned(16))) __bf16 hs[CC * HLD];  // 37.9 KB  h^T[d][p]
    int bq = blockIdx.x >> 2;
    int g  = blockIdx.x & 3;
    const float*  xg = x + (size_t)(bq * GG + g) * PP * CC;
    const __bf16* Mg = params + (size_t)bq * NGEN + g * GPARAM;          // M row-major 64x64
    const __bf16* Sg = Mg + PWSZ;                                        // S row-major 32x288
    const float*  Mb = mb + bq * NBIAS + g * (CC + OPTS);                // [0..63]
    const float*  Sb = Mb + CC;                                          // [0..31]
    int wave = threadIdx.x >> 5;
    int lane = threadIdx.x & 31;
    int hi = lane >> 4, cl = lane & 15;

    // Cooperative transpose of M into LDS: Mt[d][c] = M[c][d]
    {
        int c  = threadIdx.x >> 2;           // 0..63
        int db = (threadIdx.x & 3) * 16;     // 0,16,32,48
#pragma unroll
        for (int i = 0; i < 16; ++i) Mt[(db + i) * MLD + c] = Mg[c * CC + db + i];
    }
    __syncthreads();

    // GEMM1: 18x4 = 72 tiles of 16x16, 9 per wave, K = 64 (2 steps)
#pragma unroll
    for (int i = 0; i < 9; ++i) {
        int t = wave * 9 + i;
        int tp = t >> 2, td = t & 3;
        v8f acc = {};
#pragma unroll
        for (int kk = 0; kk < 2; ++kk) {
            v16bf a = load_a_f32(xg + tp * 16 * CC + kk * 32, CC);
            v16bf b = load_bt(Mt + td * 16 * MLD + kk * 32, MLD);
            acc = WMMA_BF16(acc, a, b);
        }
        int d = td * 16 + cl;
        float mbv = Mb[d];
        v8bf hv;
#pragma unroll
        for (int v = 0; v < 8; ++v) {
            float val = acc[v] + mbv;
            hv[v] = (__bf16)(val > 0.0f ? val : 0.0f);
        }
        *(v8bf*)(hs + d * HLD + tp * 16 + 8 * hi) = hv;  // one 16B LDS store
    }
    __syncthreads();

    // GEMM2: 2x4 = 8 tiles, one per wave, K = 288 (9 steps), B = h^T from LDS
    int to = wave >> 2, td = wave & 3;
    v8f acc = {};
#pragma unroll
    for (int kk = 0; kk < 9; ++kk) {
        v16bf a = load_a_bf(Sg + to * 16 * PP + kk * 32, PP);
        v16bf b = load_bt(hs + td * 16 * HLD + kk * 32, HLD);
        acc = WMMA_BF16(acc, a, b);
    }
    int d = td * 16 + cl;
    __bf16* outg = h2 + (size_t)bq * HFLAT + g * CC + d;
#pragma unroll
    for (int v = 0; v < 8; ++v) {
        int o = to * 16 + v + 8 * hi;
        float val = acc[v] + Sb[o];
        outg[o * (GG * CC)] = (__bf16)(val > 0.0f ? val : 0.0f);
    }
}

// ---------------- Stage 4: W_v GEMM (1200 x 256, K=8192), 4-way K-split ----------------
// Each wave: one 16x16 tile over K-range of 2048; fp32 partial to ws.
__global__ __launch_bounds__(256) void out_gemm_kernel(const __bf16* __restrict__ h2,
                                                       const __bf16* __restrict__ wvT,
                                                       float* __restrict__ part) {
    const int tiles_n = ODIM / 16;                     // 16
    const int kchunk  = HFLAT / KSPLIT;                // 2048
    int id = blockIdx.x * 8 + (threadIdx.x >> 5);      // 0..4799
    int ks = id / (75 * tiles_n);
    int r  = id % (75 * tiles_n);
    int tm = r / tiles_n, tn = r % tiles_n;
    const __bf16* A  = h2  + (size_t)tm * 16 * HFLAT + ks * kchunk;
    const __bf16* Bt = wvT + (size_t)tn * 16 * HFLAT + ks * kchunk;
    v8f acc = {};
    for (int kk = 0; kk < kchunk / 32; ++kk) {
        v16bf a = load_a_bf(A + kk * 32, HFLAT);
        v16bf b = load_bt(Bt + kk * 32, HFLAT);
        acc = WMMA_BF16(acc, a, b);
    }
    int lane = threadIdx.x & 31;
    int col = tn * 16 + (lane & 15);
    int hi = lane >> 4;
    float* dst = part + (size_t)ks * BQ * ODIM;
#pragma unroll
    for (int v = 0; v < 8; ++v) {
        int row = tm * 16 + v + 8 * hi;
        dst[row * ODIM + col] = acc[v];
    }
}

// ---------------- Stage 5: K-split reduce + bias + LayerNorm over 256 ----------------
__global__ __launch_bounds__(256) void ln_kernel(const float* __restrict__ part,
                                                 const float* __restrict__ bv,
                                                 const float* __restrict__ g,
                                                 const float* __restrict__ b,
                                                 float* __restrict__ out) {
    __shared__ float s1[256], s2[256];
    int row = blockIdx.x, t = threadIdx.x;
    float v = bv[t];
#pragma unroll
    for (int ks = 0; ks < KSPLIT; ++ks) v += part[(size_t)ks * BQ * ODIM + row * ODIM + t];
    s1[t] = v; s2[t] = v * v;
    __syncthreads();
    for (int s = 128; s > 0; s >>= 1) {
        if (t < s) { s1[t] += s1[t + s]; s2[t] += s2[t + s]; }
        __syncthreads();
    }
    float mean = s1[0] * (1.0f / ODIM);
    float var  = s2[0] * (1.0f / ODIM) - mean * mean;
    float inv  = __frsqrt_rn(var + 1e-5f);
    out[row * ODIM + t] = (v - mean) * inv * g[t] + b[t];
}

// ---------------- launch ----------------
extern "C" void kernel_launch(void* const* d_in, const int* in_sizes, int n_in,
                              void* d_out, int out_size, void* d_ws, size_t ws_size,
                              hipStream_t stream) {
    const float* x      = (const float*)d_in[0];
    const float* query  = (const float*)d_in[1];
    const float* W_gen  = (const float*)d_in[2];
    const float* b_gen  = (const float*)d_in[3];
    const float* W_bias = (const float*)d_in[4];
    const float* b_bias = (const float*)d_in[5];
    const float* W_v    = (const float*)d_in[6];
    const float* b_v    = (const float*)d_in[7];
    const float* ln_g   = (const float*)d_in[8];
    const float* ln_b   = (const float*)d_in[9];
    float* out = (float*)d_out;

    char* ws = (char*)d_ws;
    size_t off = 0;
    auto take = [&](size_t bytes) { void* p = ws + off; off += (bytes + 255) & ~(size_t)255; return p; };
    __bf16* wgenT     = (__bf16*)take((size_t)QD * NGEN * 2);          // 27.3 MB (53248 x 256)
    __bf16* q_bf      = (__bf16*)take((size_t)BQ * QD * 2);            // 0.6 MB
    __bf16* wvT       = (__bf16*)take((size_t)HFLAT * ODIM * 2);       // 4.2 MB  (256 x 8192)
    __bf16* params_bf = (__bf16*)take((size_t)BQ * NGEN * 2);          // 127.8 MB
    float*  mb_ws     = (float*) take((size_t)BQ * NBIAS * 4);         // 1.8 MB
    __bf16* h2_bf     = (__bf16*)take((size_t)BQ * HFLAT * 2);         // 19.7 MB
    float*  part      = (float*) take((size_t)KSPLIT * BQ * ODIM * 4); // 4.9 MB

    int n;
    n = QD * NGEN;    cvt_bf16_T_kernel<<<(n + 255) / 256, 256, 0, stream>>>(W_gen, wgenT, QD, NGEN);
    n = BQ * QD;      cvt_bf16_kernel  <<<(n + 255) / 256, 256, 0, stream>>>(query, q_bf, n);
    n = HFLAT * ODIM; cvt_bf16_T_kernel<<<(n + 255) / 256, 256, 0, stream>>>(W_v, wvT, HFLAT, ODIM);

    mixbias_kernel<<<(BQ * NBIAS + 255) / 256, 256, 0, stream>>>(query, W_bias, b_bias, mb_ws);

    // 25 M-blocks x 832 N-blocks = 20800 wave-jobs, 8 per workgroup
    gen_gemm_kernel<<<(25 * 832) / 8, 256, 0, stream>>>(q_bf, wgenT, b_gen, params_bf);

    mix_kernel<<<BQ * GG, 256, 0, stream>>>(x, params_bf, mb_ws, h2_bf);

    // KSPLIT * 75 * 16 = 4800 wave-jobs, 8 per workgroup
    out_gemm_kernel<<<(KSPLIT * 75 * 16) / 8, 256, 0, stream>>>(h2_bf, wvT, part);

    ln_kernel<<<BQ, 256, 0, stream>>>(part, b_v, ln_g, ln_b, out);
}
